// Decoder_65429531787576
// MI455X (gfx1250) — matrix operand order repaired and checked by reading, not compiled, not run on hardware
//
#include <hip/hip_runtime.h>
#include <hip/hip_bf16.h>
#include <math.h>

// ---------------------------------------------------------------------------
// Problem constants (from reference)
// ---------------------------------------------------------------------------
#define BB   64
#define SS   2048
#define HH   512
#define EE   256
#define VV   32000
#define LL   2
#define OOVN 50

typedef __attribute__((ext_vector_type(16))) __bf16 v16bf;
typedef __attribute__((ext_vector_type(8)))  float  v8f;

// ---------------------------------------------------------------------------
// WMMA fragment layouts used by this code (contiguous k, not the ISA table order; 16-bit, wave32):
//  A (16x32): lanes 0-15 rows M=0..15 K=0..15; lanes 16-31 same
//             rows, K=16..31. VGPR i: K=(lane/16)*16+2i (+0,+1).
//  B (32x16) from row-major W[N,K] (B=W^T): lane -> column n=lane&15;
//             VGPR i: K=(lane/16)*16 + 2i (+0,+1).
// ---------------------------------------------------------------------------
__device__ __forceinline__ v16bf load_a_frag(const float* __restrict__ A,
                                             int lda, int m0, int k0) {
  const int lane = threadIdx.x & 31;
  const int m    = m0 + (lane & 15);
  const float* r = A + (size_t)m * lda + k0 + ((lane >> 4) << 4);
  const float2* r2 = reinterpret_cast<const float2*>(r);  // 8B aligned
  v16bf a;
#pragma unroll
  for (int i = 0; i < 8; ++i) {
    const int g = i >> 2, p = i & 3;
    const float2 x = r2[i];                               // K = i*2
    a[2 * i]     = (__bf16)x.x;
    a[2 * i + 1] = (__bf16)x.y;
  }
  return a;
}

__device__ __forceinline__ v16bf load_b_frag(const float* __restrict__ W,
                                             int ldw, int n0, int k0) {
  const int lane = threadIdx.x & 31;
  const int n    = n0 + (lane & 15);
  const float* r = W + (size_t)n * ldw + k0 + ((lane >> 4) << 4);
  const float4* r4 = reinterpret_cast<const float4*>(r);  // 16B aligned
  v16bf b;
#pragma unroll
  for (int i = 0; i < 4; ++i) {
    const float4 x = r4[i];
    b[4 * i]     = (__bf16)x.x;
    b[4 * i + 1] = (__bf16)x.y;
    b[4 * i + 2] = (__bf16)x.z;
    b[4 * i + 3] = (__bf16)x.w;
  }
  return b;
}

__device__ __forceinline__ v8f wmma_bf16(v16bf a, v16bf b, v8f c) {
  return __builtin_amdgcn_wmma_f32_16x16x32_bf16(
      /*neg_a=*/false, a, /*neg_b=*/false, b,
      /*c_mod=*/(short)0, c, /*reuse_a=*/false, /*reuse_b=*/false);
}

__device__ __forceinline__ float sigmoidf_(float x) {
  return 1.0f / (1.0f + __expf(-x));
}

// ---------------------------------------------------------------------------
// M=64-specialized WMMA GEMM: C[64,N] = act(A[64,K] @ W[N,K]^T + bias[N]).
// One wave per 16-column tile; each B fragment reused across 4 M-tiles.
// N multiple of 16, K multiple of 32. ACT: 0 none, 1 relu.
// ---------------------------------------------------------------------------
template <int ACT>
__global__ void gemm_wmma_m64_kernel(const float* __restrict__ A, int lda,
                                     const float* __restrict__ W, int ldw,
                                     const float* __restrict__ bias,
                                     float* __restrict__ C, int ldc,
                                     int N, int K) {
  const int wid = blockIdx.x * (blockDim.x >> 5) + (threadIdx.x >> 5);
  const int n0  = wid << 4;
  if (n0 >= N) return;                       // uniform per-wave

  v8f acc[4] = {};
  for (int k0 = 0; k0 < K; k0 += 32) {
    const v16bf b = load_b_frag(W, ldw, n0, k0);
#pragma unroll
    for (int mt = 0; mt < 4; ++mt)
      acc[mt] = wmma_bf16(load_a_frag(A, lda, mt * 16, k0), b, acc[mt]);
  }

  const int lane = threadIdx.x & 31;
  const int n    = n0 + (lane & 15);
  const int half = lane >> 4;
  const float bv = bias ? bias[n] : 0.0f;
#pragma unroll
  for (int mt = 0; mt < 4; ++mt)
#pragma unroll
    for (int j = 0; j < 8; ++j) {
      float v = acc[mt][j] + bv;
      if (ACT == 1) v = fmaxf(v, 0.0f);
      C[(size_t)(mt * 16 + j + 8 * half) * ldc + n] = v;
    }
}

// ---------------------------------------------------------------------------
// Pre-pass: convert W_ah (fp32 [H,H]) into bf16 B-fragments, laid out
// [nt 0..31][kt 0..15][lane 0..31][16 bf16] so the attention inner loop
// reads one fully coalesced 32B chunk per lane.
// ---------------------------------------------------------------------------
__global__ void wah_swizzle_kernel(const float* __restrict__ W,
                                   __bf16* __restrict__ Bfrag) {
  const int t    = blockIdx.x * blockDim.x + threadIdx.x;  // 32*16*32 threads
  const int lane = t & 31;
  const int ktnt = t >> 5;
  const int kt   = ktnt & 15;
  const int nt   = ktnt >> 4;
  const int n    = nt * 16 + (lane & 15);
  const int k0   = kt * 32 + ((lane >> 4) << 4);
  const float4* r4 =
      reinterpret_cast<const float4*>(W + (size_t)n * HH + k0);
  v16bf b;
#pragma unroll
  for (int i = 0; i < 4; ++i) {
    const float4 x = r4[i];
    b[4 * i]     = (__bf16)x.x;
    b[4 * i + 1] = (__bf16)x.y;
    b[4 * i + 2] = (__bf16)x.z;
    b[4 * i + 3] = (__bf16)x.w;
  }
  *reinterpret_cast<v16bf*>(Bfrag + (size_t)t * 16) = b;
}

// ---------------------------------------------------------------------------
// Fused attention logits (68.7 GFLOP hot spot):
//   e[b,s] = sum_n tanh( enc[b,s,:]@W_ah.T[n] + dec_feat[b,n] + b_ah[n]
//                        + coverage[b,s]*W_ac[n] + b_ac[n] ) * W_av[n]
// (b_av dropped: softmax is shift-invariant.)
//
// Block = 256 threads (8 waves) per 64 flat rows of [B*S, H]:
//   stage:  the block cooperatively converts the 64x512 row tile to bf16
//           WMMA A-fragments in LDS ([mt][kt][lane][32B] = 64 KB); enc is
//           read from HBM exactly once.
//   compute: wave w owns 2 M-tiles (rows 32*(w/4)..+32, A = 32 fragments
//           ~256 VGPRs, promoted to registers from LDS) and an 8-tile
//           n-slice (w%4). Per n-tile all 16 B fragments are prefetched
//           (32 global_load_b128 in flight) then consumed by 2 WMMAs each.
//           Total ~450 VGPRs leaves the allocator room to keep the B
//           pipeline live (the 4-Mtile variant at ~700 VGPRs serialized it).
//   reduce: lane-shuffle over 16 columns, then 4-way cross-wave (n-slices)
//           via 1 KB of reused LDS.
// ---------------------------------------------------------------------------
__global__ __launch_bounds__(256) void attn_logits_kernel(
    const float* __restrict__ enc,       // [B*S, H]
    const __bf16* __restrict__ Bfrag,    // swizzled W_ah
    const float* __restrict__ dec_feat,  // [B,H]
    const float* __restrict__ b_ah,      // [H]
    const float* __restrict__ b_ac,      // [H]
    const float* __restrict__ W_ac,      // [H]
    const float* __restrict__ W_av,      // [H]
    const float* __restrict__ coverage,  // [B*S]
    float* __restrict__ e_out) {         // [B*S]
  __shared__ __align__(32) char smraw[4 * 16 * 32 * 32];   // 65536 B
  __bf16* lds = reinterpret_cast<__bf16*>(smraw);

  const int wv   = threadIdx.x >> 5;           // wave in block: 0..7
  const int lane = threadIdx.x & 31;
  const int half = lane >> 4;
  const int m0b  = blockIdx.x * 64;            // 64 | S -> one batch row
  const int b    = m0b / SS;

  // ---- stage A tile: wave w swizzles M-tile w/2, kt-half w%2 ----
  {
    const int mts    = wv >> 1;
    const int ktbase = (wv & 1) * 8;
#pragma unroll
    for (int kti = 0; kti < 8; ++kti) {
      const int kt  = ktbase + kti;
      const v16bf a = load_a_frag(enc, HH, m0b + mts * 16, kt * 32);
      *reinterpret_cast<v16bf*>(&lds[((size_t)(mts * 16 + kt) * 32 + lane) * 16]) = a;
    }
  }
  __syncthreads();

  const int mtp = wv >> 2;                     // 0: rows 0..31, 1: rows 32..63
  const int nsl = wv & 3;                      // n-slice: n-tiles [8*nsl, +8)

  // ---- pull this wave's 32 A fragments into registers (~256 VGPRs) ----
  v16bf af[2][16];
#pragma unroll
  for (int mi = 0; mi < 2; ++mi)
#pragma unroll
    for (int kt = 0; kt < 16; ++kt)
      af[mi][kt] = *reinterpret_cast<const v16bf*>(
          &lds[((size_t)((mtp * 2 + mi) * 16 + kt) * 32 + lane) * 16]);

  // per-row coverage (constant across n)
  float covj[2][8];
#pragma unroll
  for (int mi = 0; mi < 2; ++mi)
#pragma unroll
    for (int j = 0; j < 8; ++j)
      covj[mi][j] = coverage[m0b + mtp * 32 + mi * 16 + j + 8 * half];

  float ep[2][8];
#pragma unroll
  for (int mi = 0; mi < 2; ++mi)
#pragma unroll
    for (int j = 0; j < 8; ++j) ep[mi][j] = 0.0f;

  // ---- compute: 8 n-tiles x 16 kt x 2 M-tiles ----
  for (int nti = 0; nti < 8; ++nti) {
    const int nt = nsl * 8 + nti;
    const int n  = nt * 16 + (lane & 15);
    const float cadd = dec_feat[b * HH + n] + b_ah[n] + b_ac[n];
    const float wac  = W_ac[n];
    const float wav  = W_av[n];

    // issue all 16 B-fragment loads for this n-tile up front (pipelined)
    v16bf bf[16];
#pragma unroll
    for (int kt = 0; kt < 16; ++kt)
      bf[kt] = *reinterpret_cast<const v16bf*>(
          Bfrag + ((size_t)(nt * 16 + kt) * 32 + lane) * 16);

    v8f acc[2] = {};
#pragma unroll
    for (int kt = 0; kt < 16; ++kt) {
#pragma unroll
      for (int mi = 0; mi < 2; ++mi)
        acc[mi] = wmma_bf16(af[mi][kt], bf[kt], acc[mi]);
    }
#pragma unroll
    for (int mi = 0; mi < 2; ++mi)
#pragma unroll
      for (int j = 0; j < 8; ++j)
        ep[mi][j] += tanhf(acc[mi][j] + cadd + covj[mi][j] * wac) * wav;
  }

  // ---- reduce over the 16 columns held by each lane half ----
#pragma unroll
  for (int mi = 0; mi < 2; ++mi)
#pragma unroll
    for (int j = 0; j < 8; ++j) {
      ep[mi][j] += __shfl_xor(ep[mi][j], 8, 32);
      ep[mi][j] += __shfl_xor(ep[mi][j], 4, 32);
      ep[mi][j] += __shfl_xor(ep[mi][j], 2, 32);
      ep[mi][j] += __shfl_xor(ep[mi][j], 1, 32);
    }

  // ---- cross-wave reduce over the 4 n-slices via reused LDS ----
  __syncthreads();                      // everyone done reading A fragments
  float* red = reinterpret_cast<float*>(smraw);   // [4 n-slices][64 rows]
  if ((lane & 15) == 0) {
#pragma unroll
    for (int mi = 0; mi < 2; ++mi)
#pragma unroll
      for (int j = 0; j < 8; ++j)
        red[nsl * 64 + mtp * 32 + mi * 16 + j + 8 * half] = ep[mi][j];
  }
  __syncthreads();
  if (threadIdx.x < 64) {
    const float s = red[threadIdx.x] + red[64 + threadIdx.x] +
                    red[128 + threadIdx.x] + red[192 + threadIdx.x];
    e_out[m0b + threadIdx.x] = s;
  }
}

// ---------------------------------------------------------------------------
// Elementwise / reduction kernels
// ---------------------------------------------------------------------------
__global__ void embed_kernel(const int* __restrict__ input,
                             const float* __restrict__ emb,
                             float* __restrict__ out) {      // [B,E]
  const int idx = blockIdx.x * blockDim.x + threadIdx.x;     // B*E
  const int b = idx / EE, e = idx % EE;
  out[idx] = emb[(size_t)input[b] * EE + e];
}

__global__ void concat2_kernel(const float* __restrict__ X, int nx,
                               const float* __restrict__ Y, int ny,
                               float* __restrict__ O) {      // [B, nx+ny]
  const int n   = nx + ny;
  const int idx = blockIdx.x * blockDim.x + threadIdx.x;     // B*n
  const int b = idx / n, c = idx % n;
  O[idx] = (c < nx) ? X[(size_t)b * nx + c] : Y[(size_t)b * ny + (c - nx)];
}

// softmax over S per batch row; also emits sum_coverage = coverage + a_t
__global__ void attn_softmax_kernel(const float* __restrict__ e,
                                    const float* __restrict__ coverage,
                                    float* __restrict__ a_t,
                                    float* __restrict__ sum_cov) {
  __shared__ float red[256];
  const int b = blockIdx.x, tid = threadIdx.x;
  const float* row = e + (size_t)b * SS;
  float mx = -3.4e38f;
  for (int s = tid; s < SS; s += 256) mx = fmaxf(mx, row[s]);
  red[tid] = mx; __syncthreads();
  for (int st = 128; st > 0; st >>= 1) {
    if (tid < st) red[tid] = fmaxf(red[tid], red[tid + st]);
    __syncthreads();
  }
  mx = red[0]; __syncthreads();
  float sm = 0.0f;
  for (int s = tid; s < SS; s += 256) sm += __expf(row[s] - mx);
  red[tid] = sm; __syncthreads();
  for (int st = 128; st > 0; st >>= 1) {
    if (tid < st) red[tid] += red[tid + st];
    __syncthreads();
  }
  const float inv = 1.0f / red[0];
  for (int s = tid; s < SS; s += 256) {
    const float a = __expf(row[s] - mx) * inv;
    a_t[(size_t)b * SS + s]     = a;
    sum_cov[(size_t)b * SS + s] = coverage[(size_t)b * SS + s] + a;
  }
}

// context[b,h] = sum_s a_t[b,s] * enc[b,s,h]  (pure bandwidth, 256 MB read)
__global__ void context_kernel(const float* __restrict__ a_t,
                               const float* __restrict__ enc,
                               float* __restrict__ ctx) {
  const int b = blockIdx.x >> 1;
  const int h = ((blockIdx.x & 1) << 8) + threadIdx.x;       // H = 2*256
  const float* arow = a_t + (size_t)b * SS;
  const float* e    = enc + (size_t)b * SS * HH + h;
  float acc = 0.0f;
  for (int s = 0; s < SS; ++s) acc += arow[s] * e[(size_t)s * HH];
  ctx[(size_t)b * HH + h] = acc;
}

// GRU gate combine (biases already folded into gi/gh by the GEMMs)
__global__ void gru_gate_kernel(const float* __restrict__ gi,   // [B,3H]
                                const float* __restrict__ gh,   // [B,3H]
                                const float* __restrict__ hprev,// [B,H]
                                float* __restrict__ hout) {     // [B,H]
  const int idx = blockIdx.x * blockDim.x + threadIdx.x;        // B*H
  const int b = idx / HH, h = idx % HH;
  const float* gib = gi + (size_t)b * 3 * HH;
  const float* ghb = gh + (size_t)b * 3 * HH;
  const float r  = sigmoidf_(gib[h]      + ghb[h]);
  const float z  = sigmoidf_(gib[HH + h] + ghb[HH + h]);
  const float nn = tanhf(gib[2 * HH + h] + r * ghb[2 * HH + h]);
  hout[idx] = (1.0f - z) * nn + z * hprev[idx];
}

__global__ void pgen_kernel(const float* __restrict__ ctx,    // [B,H]
                            const float* __restrict__ h1,     // [B,H]
                            const float* __restrict__ gru_in, // [B,E]
                            const float* __restrict__ W_ph, const float* __restrict__ b_ph,
                            const float* __restrict__ W_ps, const float* __restrict__ b_ps,
                            const float* __restrict__ W_px, const float* __restrict__ b_px,
                            float* __restrict__ pgen) {
  __shared__ float red[256];
  const int b = blockIdx.x, tid = threadIdx.x;
  float s = 0.0f;
  for (int i = tid; i < HH; i += 256)
    s += ctx[(size_t)b * HH + i] * W_ph[i] + h1[(size_t)b * HH + i] * W_ps[i];
  for (int i = tid; i < EE; i += 256)
    s += gru_in[(size_t)b * EE + i] * W_px[i];
  red[tid] = s; __syncthreads();
  for (int st = 128; st > 0; st >>= 1) {
    if (tid < st) red[tid] += red[tid + st];
    __syncthreads();
  }
  if (tid == 0)
    pgen[b] = sigmoidf_(red[0] + b_ph[0] + b_ps[0] + b_px[0]);
}

// vocab softmax * pgen -> final[:, :V]; zero the OOV tail
__global__ void vocab_finalize_kernel(const float* __restrict__ logits, // [B,V]
                                      const float* __restrict__ pgen,
                                      float* __restrict__ out) {        // [B,V+OOV]
  __shared__ float red[256];
  const int b = blockIdx.x, tid = threadIdx.x;
  const float* row = logits + (size_t)b * VV;
  float mx = -3.4e38f;
  for (int v = tid; v < VV; v += 256) mx = fmaxf(mx, row[v]);
  red[tid] = mx; __syncthreads();
  for (int st = 128; st > 0; st >>= 1) {
    if (tid < st) red[tid] = fmaxf(red[tid], red[tid + st]);
    __syncthreads();
  }
  mx = red[0]; __syncthreads();
  float sm = 0.0f;
  for (int v = tid; v < VV; v += 256) sm += __expf(row[v] - mx);
  red[tid] = sm; __syncthreads();
  for (int st = 128; st > 0; st >>= 1) {
    if (tid < st) red[tid] += red[tid + st];
    __syncthreads();
  }
  const float scale = pgen[b] / red[0];
  float* orow = out + (size_t)b * (VV + OOVN);
  for (int v = tid; v < VV + OOVN; v += 256)
    orow[v] = (v < VV) ? __expf(row[v] - mx) * scale : 0.0f;
}

// pointer scatter: final[b, ewo[b,s]] += a_t[b,s] * (1 - pgen[b])
__global__ void scatter_kernel(const int* __restrict__ ewo,   // [B,S]
                               const float* __restrict__ a_t, // [B,S]
                               const float* __restrict__ pgen,
                               float* __restrict__ out) {     // [B,V+OOV]
  const int idx = blockIdx.x * blockDim.x + threadIdx.x;      // B*S
  const int b = idx / SS;
  const float p = a_t[idx] * (1.0f - pgen[b]);
  atomicAdd(out + (size_t)b * (VV + OOVN) + ewo[idx], p);
}

// ---------------------------------------------------------------------------
// Host launch
// ---------------------------------------------------------------------------
extern "C" void kernel_launch(void* const* d_in, const int* in_sizes, int n_in,
                              void* d_out, int out_size, void* d_ws, size_t ws_size,
                              hipStream_t stream) {
  (void)in_sizes; (void)n_in; (void)out_size; (void)ws_size;
  // inputs (setup_inputs order)
  const int*   input    = (const int*)  d_in[0];
  const float* hidden   = (const float*)d_in[1];   // [L,B,H]
  const float* enc      = (const float*)d_in[2];   // [B,S,H]
  const float* ctx_vec  = (const float*)d_in[3];   // [B,H]
  /* oovs_zero d_in[4] unused (we zero the tail ourselves) */
  const int*   ewo      = (const int*)  d_in[5];   // [B,S]
  const float* coverage = (const float*)d_in[6];   // [B,S]
  const float* emb      = (const float*)d_in[7];
  const float* W_ah = (const float*)d_in[8],  *b_ah = (const float*)d_in[9];
  const float* W_as = (const float*)d_in[10], *b_as = (const float*)d_in[11];
  const float* W_ac = (const float*)d_in[12], *b_ac = (const float*)d_in[13];
  const float* W_av = (const float*)d_in[14]; /* b_av d_in[15]: softmax-invariant */
  const float* W_gin = (const float*)d_in[16], *b_gin = (const float*)d_in[17];
  const float* W_ih0 = (const float*)d_in[18], *W_hh0 = (const float*)d_in[19];
  const float* b_ih0 = (const float*)d_in[20], *b_hh0 = (const float*)d_in[21];
  const float* W_ih1 = (const float*)d_in[22], *W_hh1 = (const float*)d_in[23];
  const float* b_ih1 = (const float*)d_in[24], *b_hh1 = (const float*)d_in[25];
  const float* W_ph = (const float*)d_in[26], *b_ph = (const float*)d_in[27];
  const float* W_ps = (const float*)d_in[28], *b_ps = (const float*)d_in[29];
  const float* W_px = (const float*)d_in[30], *b_px = (const float*)d_in[31];
  const float* W_f1 = (const float*)d_in[32], *b_f1 = (const float*)d_in[33];
  const float* W_f2 = (const float*)d_in[34], *b_f2 = (const float*)d_in[35];

  // output regions (return order, flat)
  float* out      = (float*)d_out;
  float* fw       = out;                                   // [B, V+OOV]
  float* nh       = out + (size_t)BB * (VV + OOVN);        // [L,B,H]
  float* ctx_out  = nh  + (size_t)LL * BB * HH;            // [B,H]
  float* a_t      = ctx_out + (size_t)BB * HH;             // [B,S]
  float* sum_cov  = a_t + (size_t)BB * SS;                 // [B,S]
  float* h0 = nh;
  float* h1 = nh + (size_t)BB * HH;

  // workspace layout (floats; all region starts 32B aligned)
  float* ws        = (float*)d_ws;
  float* dec_feat  = ws;                         // 64*512
  float* e_t       = dec_feat + BB * HH;         // 64*2048
  float* embedded  = e_t + BB * SS;              // 64*256
  float* concat_ce = embedded + BB * EE;         // 64*768
  float* gru_in    = concat_ce + BB * (HH + EE); // 64*256
  float* gi        = gru_in + BB * EE;           // 64*1536
  float* gh        = gi + BB * 3 * HH;           // 64*1536
  float* dec_ctx   = gh + BB * 3 * HH;           // 64*1024
  float* fhid      = dec_ctx + BB * 2 * HH;      // 64*512
  float* logits    = fhid + BB * HH;             // 64*32000
  float* pgen      = logits + (size_t)BB * VV;   // 64
  __bf16* wah_bf   = (__bf16*)(pgen + 64);       // 512*512 bf16 (swizzled)

  const dim3 blk(256);

  // 0) one-time bf16 swizzle of W_ah into B-fragment layout
  wah_swizzle_kernel<<<dim3((32 * 16 * 32) / 256), blk, 0, stream>>>(W_ah, wah_bf);

  // 1) embedding gather + decoder input concat
  embed_kernel<<<dim3((BB * EE) / 256), blk, 0, stream>>>(input, emb, embedded);
  concat2_kernel<<<dim3((BB * (HH + EE)) / 256), blk, 0, stream>>>(
      ctx_vec, HH, embedded, EE, concat_ce);

  // 2) dec_feat = hidden[-1] @ W_as.T + b_as          (32 n-tiles)
  gemm_wmma_m64_kernel<0><<<dim3(4), blk, 0, stream>>>(
      hidden + (size_t)BB * HH, HH, W_as, HH, b_as, dec_feat, HH, HH, HH);

  // 3) fused attention logits over [B*S, H]           (2048 blocks x 8 waves)
  attn_logits_kernel<<<dim3(BB * SS / 64), blk, 0, stream>>>(
      enc, wah_bf, dec_feat, b_ah, b_ac, W_ac, W_av, coverage, e_t);

  // 4) softmax over S -> a_t, sum_coverage
  attn_softmax_kernel<<<dim3(BB), blk, 0, stream>>>(e_t, coverage, a_t, sum_cov);

  // 5) context = a_t @ enc
  context_kernel<<<dim3(BB * 2), blk, 0, stream>>>(a_t, enc, ctx_out);

  // 6) gru_in = concat(ctx_vec, embedded) @ W_gin.T + b_gin   (K=768)
  gemm_wmma_m64_kernel<0><<<dim3(2), blk, 0, stream>>>(
      concat_ce, HH + EE, W_gin, HH + EE, b_gin, gru_in, EE, EE, HH + EE);

  // 7) GRU layer 0
  gemm_wmma_m64_kernel<0><<<dim3(12), blk, 0, stream>>>(
      gru_in, EE, W_ih0, EE, b_ih0, gi, 3 * HH, 3 * HH, EE);
  gemm_wmma_m64_kernel<0><<<dim3(12), blk, 0, stream>>>(
      hidden, HH, W_hh0, HH, b_hh0, gh, 3 * HH, 3 * HH, HH);
  gru_gate_kernel<<<dim3((BB * HH) / 256), blk, 0, stream>>>(gi, gh, hidden, h0);

  // 8) GRU layer 1
  gemm_wmma_m64_kernel<0><<<dim3(12), blk, 0, stream>>>(
      h0, HH, W_ih1, HH, b_ih1, gi, 3 * HH, 3 * HH, HH);
  gemm_wmma_m64_kernel<0><<<dim3(12), blk, 0, stream>>>(
      hidden + (size_t)BB * HH, HH, W_hh1, HH, b_hh1, gh, 3 * HH, 3 * HH, HH);
  gru_gate_kernel<<<dim3((BB * HH) / 256), blk, 0, stream>>>(
      gi, gh, hidden + (size_t)BB * HH, h1);

  // 9) pgen
  pgen_kernel<<<dim3(BB), blk, 0, stream>>>(
      ctx_out, h1, gru_in, W_ph, b_ph, W_ps, b_ps, W_px, b_px, pgen);

  // 10) vocab path: relu(concat(h1,ctx) @ W_f1.T + b_f1) @ W_f2.T + b_f2
  concat2_kernel<<<dim3((BB * 2 * HH) / 256), blk, 0, stream>>>(
      h1, HH, ctx_out, HH, dec_ctx);
  gemm_wmma_m64_kernel<1><<<dim3(4), blk, 0, stream>>>(
      dec_ctx, 2 * HH, W_f1, 2 * HH, b_f1, fhid, HH, HH, 2 * HH);
  gemm_wmma_m64_kernel<0><<<dim3(250), blk, 0, stream>>>(
      fhid, HH, W_f2, HH, b_f2, logits, VV, VV, HH);

  // 11) softmax over V, * pgen, zero OOV tail
  vocab_finalize_kernel<<<dim3(BB), blk, 0, stream>>>(logits, pgen, fw);

  // 12) pointer-generator scatter add
  scatter_kernel<<<dim3((BB * SS) / 256), blk, 0, stream>>>(ewo, a_t, pgen, fw);
}
